// HFMiMoV2Attention_61546881352285
// MI455X (gfx1250) — compile-verified, matrix operand order checked by reading, not run on hardware
//
#include <hip/hip_runtime.h>
#include <hip/hip_bf16.h>

typedef __attribute__((ext_vector_type(16))) __bf16 v16bf;
typedef __attribute__((ext_vector_type(8)))  __bf16 v8bf;
typedef __attribute__((ext_vector_type(8)))  float  v8f;

constexpr int S_ = 2048, HID_ = 2048;
constexpr int H_ = 16, HK_ = 4, D_ = 128, DV_ = 128;
constexpr float SCALE_ = 0.08838834764831845f; // 1/sqrt(128)

__device__ __forceinline__ v8f wmma_bf16(v16bf a, v16bf b, v8f c) {
  return __builtin_amdgcn_wmma_f32_16x16x32_bf16(false, a, false, b, (short)0, c,
                                                 false, false);
}

// Fragment for 16-bit WMMA A/B operand, per-lane: two contiguous 8-element runs
// of one row: k = hf*8 + 0..7 and k = 16 + hf*8 + 0..7  (matches ISA 7.12.2).
__device__ __forceinline__ v16bf frag_row(const __bf16* row, int hf) {
  v8bf lo = *(const v8bf*)(row + hf * 8);
  v8bf hi = *(const v8bf*)(row + 16 + hf * 8);
  v16bf f;
#pragma unroll
  for (int t = 0; t < 8; ++t) { f[t] = lo[t]; f[8 + t] = hi[t]; }
  return f;
}

// ---- CDNA5 async copy global->LDS (ASYNCcnt) --------------------------------
__device__ __forceinline__ void async_load_b128(const void* gaddr, unsigned lds_off) {
  asm volatile("global_load_async_to_lds_b128 %0, %1, off"
               :: "v"(lds_off), "v"(gaddr) : "memory");
}
__device__ __forceinline__ void wait_async0() {
  asm volatile("s_wait_asynccnt 0" ::: "memory");
}

// ---------------------------------------------------------------- f32 -> bf16
__global__ void cvt_bf16_kernel(const float* __restrict__ src,
                                __bf16* __restrict__ dst, int n) {
  int i = blockIdx.x * blockDim.x + threadIdx.x;
  if (i < n) dst[i] = (__bf16)src[i];
}

// -------------------------------------------------- f32 [K][N] -> bf16 [N][K]
__global__ void cvt_t_kernel(const float* __restrict__ src,
                             __bf16* __restrict__ dst, int K, int N) {
  __shared__ float tile[32][33];
  int kb = blockIdx.y * 32, nb = blockIdx.x * 32;
  int tx = threadIdx.x, ty = threadIdx.y;   // block (32, 8)
#pragma unroll
  for (int i = 0; i < 32; i += 8)
    tile[ty + i][tx] = src[(size_t)(kb + ty + i) * N + nb + tx];
  __syncthreads();
#pragma unroll
  for (int i = 0; i < 32; i += 8)
    dst[(size_t)(nb + ty + i) * K + kb + tx] = (__bf16)tile[tx][ty + i];
}

// ---------------------------------------------------------------- LDS-staged WMMA GEMM
// C[M,N](f32) = A[M,K](bf16, row-major) * Bt[N,K](bf16, pre-transposed)
// Block 256 threads = 8 waves; BM=128, BN=64, BK=32; 32x32 per wave (4 WMMA).
// Panels staged via global_load_async_to_lds_b128, double-buffered.
__global__ void __launch_bounds__(256)
gemm_bf16_kernel(const __bf16* __restrict__ A, const __bf16* __restrict__ Bt,
                 float* __restrict__ C, int M, int N, int K) {
  __shared__ __bf16 sA[2][128][32];
  __shared__ __bf16 sB[2][64][32];

  int tid  = threadIdx.x;
  int lane = tid & 31, wave = tid >> 5;
  int hf = lane >> 4, l15 = lane & 15;
  int tm = blockIdx.y * 128, tn = blockIdx.x * 64;
  int wm = (wave & 3) * 32, wn = (wave >> 2) * 32;

  int prow = tid >> 2;            // 0..63
  int pcol = (tid & 3) * 8;       // element offset of 16B chunk

  auto issue = [&](int bf, int kk) {
    async_load_b128(A + (size_t)(tm + prow) * K + kk + pcol,
                    (unsigned)(size_t)&sA[bf][prow][pcol]);
    async_load_b128(A + (size_t)(tm + prow + 64) * K + kk + pcol,
                    (unsigned)(size_t)&sA[bf][prow + 64][pcol]);
    async_load_b128(Bt + (size_t)(tn + prow) * K + kk + pcol,
                    (unsigned)(size_t)&sB[bf][prow][pcol]);
  };

  v8f acc[2][2] = {};
  int nk = K / 32;
  issue(0, 0);
  for (int s = 0; s < nk; ++s) {
    wait_async0();
    __syncthreads();
    if (s + 1 < nk) issue((s + 1) & 1, (s + 1) * 32);
    int bf = s & 1;
    v16bf a0 = frag_row(&sA[bf][wm + l15][0], hf);
    v16bf a1 = frag_row(&sA[bf][wm + 16 + l15][0], hf);
    v16bf b0 = frag_row(&sB[bf][wn + l15][0], hf);
    v16bf b1 = frag_row(&sB[bf][wn + 16 + l15][0], hf);
    acc[0][0] = wmma_bf16(a0, b0, acc[0][0]);
    acc[0][1] = wmma_bf16(a0, b1, acc[0][1]);
    acc[1][0] = wmma_bf16(a1, b0, acc[1][0]);
    acc[1][1] = wmma_bf16(a1, b1, acc[1][1]);
    __syncthreads();
  }
#pragma unroll
  for (int im = 0; im < 2; ++im)
#pragma unroll
    for (int in_ = 0; in_ < 2; ++in_)
#pragma unroll
      for (int r = 0; r < 8; ++r) {
        int m = tm + wm + im * 16 + r + 8 * hf;
        int n = tn + wn + in_ * 16 + l15;
        C[(size_t)m * N + n] = acc[im][in_][r];
      }
}

// ---------------------------------------------------------------- RoPE + pack
// Q: [s][h*128+d] -> qb[h][s][d];  K -> kb[hk][s][d];  V -> vt[hk][d][s] (transposed)
__global__ void rope_pack_kernel(const float* __restrict__ qf,
                                 const float* __restrict__ kf,
                                 const float* __restrict__ vf,
                                 const float* __restrict__ cosb,
                                 const float* __restrict__ sinb,
                                 __bf16* __restrict__ qb,
                                 __bf16* __restrict__ kb,
                                 __bf16* __restrict__ vt) {
  int s = blockIdx.x;
  const float* cs = cosb + (size_t)s * 64;
  const float* sn = sinb + (size_t)s * 64;

  for (int t = threadIdx.x; t < H_ * D_; t += blockDim.x) {
    int h = t >> 7, d = t & 127;
    size_t base = (size_t)s * (H_ * D_) + (h << 7);
    float v = qf[base + d], out;
    if (d < 32)       out = v * cs[d] - qf[base + d + 32] * sn[d];
    else if (d < 64)  out = v * cs[d] + qf[base + d - 32] * sn[d];
    else              out = v;
    qb[((size_t)h * S_ + s) * D_ + d] = (__bf16)out;
  }
  for (int t = threadIdx.x; t < HK_ * D_; t += blockDim.x) {
    int h = t >> 7, d = t & 127;
    size_t base = (size_t)s * (HK_ * D_) + (h << 7);
    float v = kf[base + d], out;
    if (d < 32)       out = v * cs[d] - kf[base + d + 32] * sn[d];
    else if (d < 64)  out = v * cs[d] + kf[base + d - 32] * sn[d];
    else              out = v;
    kb[((size_t)h * S_ + s) * D_ + d] = (__bf16)out;
  }
  for (int t = threadIdx.x; t < HK_ * DV_; t += blockDim.x) {
    int h = t >> 7, d = t & 127;
    vt[((size_t)h * DV_ + d) * S_ + s] =
        (__bf16)vf[(size_t)s * (HK_ * DV_) + (h << 7) + d];
  }
}

// ---------------------------------------------------------------- flash attention w/ sink
// 1 wave per (head, 16-query tile); single-wave workgroup.
__global__ void attn_kernel(const __bf16* __restrict__ qb,
                            const __bf16* __restrict__ kb,
                            const __bf16* __restrict__ vt,
                            const float* __restrict__ sink,
                            __bf16* __restrict__ aout) {
  __shared__ float sP[16][32];

  int lane = threadIdx.x;
  int tile = blockIdx.x;
  int h  = tile >> 7;               // S_/16 == 128
  int q0 = (tile & 127) << 4;
  int hk = h >> 2;
  int hf  = lane >> 4;
  int l15 = lane & 15;

  const __bf16* Q  = qb + ((size_t)h  * S_ + q0) * D_;
  const __bf16* Kp = kb + (size_t)hk * S_ * D_;
  const __bf16* Vt = vt + (size_t)hk * DV_ * S_;   // [d][s]

  v16bf qa[4];
#pragma unroll
  for (int c = 0; c < 4; ++c)
    qa[c] = frag_row(Q + (size_t)l15 * D_ + c * 32, hf);

  v8f o[8] = {};
  float mi[8], li[8];
#pragma unroll
  for (int r = 0; r < 8; ++r) { mi[r] = -INFINITY; li[r] = 0.0f; }

  int qlast = q0 + 15;
  for (int j = 0; j <= qlast; j += 32) {
    // scores S = Q * K^T (16x32), K-fragments are contiguous d-runs of K rows
    v8f c0 = {}, c1 = {};
#pragma unroll
    for (int c = 0; c < 4; ++c) {
      v16bf b0 = frag_row(Kp + (size_t)(j + l15) * D_ + c * 32, hf);
      v16bf b1 = frag_row(Kp + (size_t)(j + 16 + l15) * D_ + c * 32, hf);
      c0 = wmma_bf16(qa[c], b0, c0);
      c1 = wmma_bf16(qa[c], b1, c1);
    }
#pragma unroll
    for (int r = 0; r < 8; ++r) {
      int qrow = q0 + r + 8 * hf;
      c0[r] = (j + l15      <= qrow) ? c0[r] * SCALE_ : -INFINITY;
      c1[r] = (j + 16 + l15 <= qrow) ? c1[r] * SCALE_ : -INFINITY;
    }
#pragma unroll
    for (int r = 0; r < 8; ++r) {
      float rmax = fmaxf(c0[r], c1[r]);
#pragma unroll
      for (int mm = 8; mm >= 1; mm >>= 1)
        rmax = fmaxf(rmax, __shfl_xor(rmax, mm, 32));
      float mn = fmaxf(mi[r], rmax);
      float sc = __expf(mi[r] - mn);
      float p0 = __expf(c0[r] - mn);
      float p1 = __expf(c1[r] - mn);
      c0[r] = p0; c1[r] = p1;
      float rs = p0 + p1;
#pragma unroll
      for (int mm = 8; mm >= 1; mm >>= 1)
        rs += __shfl_xor(rs, mm, 32);
      li[r] = li[r] * sc + rs;
      mi[r] = mn;
#pragma unroll
      for (int c = 0; c < 8; ++c) o[c][r] *= sc;
    }
    // transpose P (C-layout -> A-fragment layout) via LDS
#pragma unroll
    for (int r = 0; r < 8; ++r) {
      sP[r + 8 * hf][l15]      = c0[r];
      sP[r + 8 * hf][16 + l15] = c1[r];
    }
    __syncthreads();
    v16bf pa;
#pragma unroll
    for (int t = 0; t < 8; ++t) {
      pa[t]     = (__bf16)sP[l15][hf * 8 + t];
      pa[8 + t] = (__bf16)sP[l15][16 + hf * 8 + t];
    }
    __syncthreads();
    // O += P * V : V^T rows give contiguous key-runs
#pragma unroll
    for (int c = 0; c < 8; ++c) {
      v16bf vfrag = frag_row(Vt + (size_t)(c * 16 + l15) * S_ + j, hf);
      o[c] = wmma_bf16(pa, vfrag, o[c]);
    }
  }

  float snk = sink[h];
#pragma unroll
  for (int r = 0; r < 8; ++r) {
    float inv = 1.0f / (li[r] + __expf(snk - mi[r]));
    int m = q0 + r + 8 * hf;
#pragma unroll
    for (int c = 0; c < 8; ++c)
      aout[(size_t)m * (H_ * DV_) + h * DV_ + c * 16 + l15] =
          (__bf16)(o[c][r] * inv);
  }
}

// ---------------------------------------------------------------- launch
extern "C" void kernel_launch(void* const* d_in, const int* in_sizes, int n_in,
                              void* d_out, int out_size, void* d_ws, size_t ws_size,
                              hipStream_t stream) {
  const float* hs   = (const float*)d_in[0];
  const float* cosb = (const float*)d_in[1];
  const float* sinb = (const float*)d_in[2];
  const float* Wq   = (const float*)d_in[3];
  const float* Wk   = (const float*)d_in[4];
  const float* Wv   = (const float*)d_in[5];
  const float* Wo   = (const float*)d_in[6];
  const float* sink = (const float*)d_in[7];
  float* out = (float*)d_out;

  char* p = (char*)d_ws;
  size_t off = 0;
  auto alloc = [&](size_t bytes) -> char* {
    char* r = p + off;
    off += (bytes + 255) & ~(size_t)255;
    return r;
  };
  __bf16* hsb  = (__bf16*)alloc((size_t)S_ * HID_ * 2);
  __bf16* wqt  = (__bf16*)alloc((size_t)HID_ * H_ * D_ * 2);    // [N=2048][K=2048]
  __bf16* wkt  = (__bf16*)alloc((size_t)HID_ * HK_ * D_ * 2);   // [512][2048]
  __bf16* wvt  = (__bf16*)alloc((size_t)HID_ * HK_ * DV_ * 2);  // [512][2048]
  __bf16* wot  = (__bf16*)alloc((size_t)H_ * DV_ * HID_ * 2);   // [2048][2048]
  float*  qf   = (float*) alloc((size_t)S_ * H_ * D_ * 4);
  float*  kf   = (float*) alloc((size_t)S_ * HK_ * D_ * 4);
  float*  vf   = (float*) alloc((size_t)S_ * HK_ * DV_ * 4);
  __bf16* qb   = (__bf16*)alloc((size_t)H_ * S_ * D_ * 2);
  __bf16* kbb  = (__bf16*)alloc((size_t)HK_ * S_ * D_ * 2);
  __bf16* vtb  = (__bf16*)alloc((size_t)HK_ * DV_ * S_ * 2);
  __bf16* aout = (__bf16*)alloc((size_t)S_ * H_ * DV_ * 2);

  cvt_bf16_kernel<<<(S_ * HID_ + 255) / 256, 256, 0, stream>>>(hs, hsb, S_ * HID_);

  dim3 tblk(32, 8);
  cvt_t_kernel<<<dim3((H_ * D_) / 32, HID_ / 32),  tblk, 0, stream>>>(Wq, wqt, HID_, H_ * D_);
  cvt_t_kernel<<<dim3((HK_ * D_) / 32, HID_ / 32), tblk, 0, stream>>>(Wk, wkt, HID_, HK_ * D_);
  cvt_t_kernel<<<dim3((HK_ * DV_) / 32, HID_ / 32),tblk, 0, stream>>>(Wv, wvt, HID_, HK_ * DV_);
  cvt_t_kernel<<<dim3(HID_ / 32, (H_ * DV_) / 32), tblk, 0, stream>>>(Wo, wot, H_ * DV_, HID_);

  // QKV projections (LDS-staged async WMMA GEMM)
  gemm_bf16_kernel<<<dim3((H_ * D_) / 64,  S_ / 128), 256, 0, stream>>>(hsb, wqt, qf, S_, H_ * D_,  HID_);
  gemm_bf16_kernel<<<dim3((HK_ * D_) / 64, S_ / 128), 256, 0, stream>>>(hsb, wkt, kf, S_, HK_ * D_, HID_);
  gemm_bf16_kernel<<<dim3((HK_ * DV_) / 64,S_ / 128), 256, 0, stream>>>(hsb, wvt, vf, S_, HK_ * DV_, HID_);

  rope_pack_kernel<<<S_, 256, 0, stream>>>(qf, kf, vf, cosb, sinb, qb, kbb, vtb);

  attn_kernel<<<H_ * (S_ / 16), 32, 0, stream>>>(qb, kbb, vtb, sink, aout);

  gemm_bf16_kernel<<<dim3(HID_ / 64, S_ / 128), 256, 0, stream>>>(aout, wot, out, S_, HID_, H_ * DV_);
}
